// GRUPolicy_64699387347566
// MI455X (gfx1250) — matrix operand order repaired and checked by reading, not compiled, not run on hardware
//
#include <hip/hip_runtime.h>
#include <math.h>

typedef __attribute__((ext_vector_type(16))) _Float16 v16h;
typedef __attribute__((ext_vector_type(8)))  _Float16 v8h;
typedef __attribute__((ext_vector_type(8)))  float    v8f;
typedef __attribute__((ext_vector_type(4)))  float    v4f;

#define OBS 48
#define HID 16
#define BATCH 2048
#define TLEN 256
#define H1 128
#define H2 128
#define ACTD 12

__device__ __forceinline__ v8f wmma_f16(v16h a, v16h b, v8f c) {
    // D = A(16x32 f16) x B(32x16 f16) + C(16x16 f32)
    return __builtin_amdgcn_wmma_f32_16x16x32_f16(false, a, false, b, (short)0, c, false, false);
}

// fast nonlinearities: v_exp_f32 + v_rcp_f32 (no IEEE div sequence)
__device__ __forceinline__ float sigmoidf_fast(float x) {
    return __builtin_amdgcn_rcpf(1.0f + __expf(-x));
}
__device__ __forceinline__ float tanhf_fast(float x) {
    float e = __expf(-2.0f * fabsf(x));
    float t = (1.0f - e) * __builtin_amdgcn_rcpf(1.0f + e);
    return copysignf(t, x);
}

// ---------------------------------------------------------------------------
// Kernel 1: GRU scan. One wave (32 threads) per 16-row batch tile.
// h kept in f32 in WMMA C-layout; per step transposed to f16 A-layout via LDS.
// Step body ordered so all h-independent work (x loads/cvt + 6 x-WMMAs) sits
// ahead of the recurrence chain (LDS row load -> 3 h-WMMAs -> gate math).
// h_seq emitted as one b128 store/lane/step from the LDS row-major view.
// ---------------------------------------------------------------------------
__global__ __launch_bounds__(32) void gru_scan_kernel(
    const float* __restrict__ x,      // [B,T,OBS]
    const float* __restrict__ h0,     // [B,HID]
    const float* __restrict__ W_ih,   // [48,48]
    const float* __restrict__ W_hh,   // [48,16]
    const float* __restrict__ b_ih,   // [48]
    const float* __restrict__ b_hh,   // [48]
    _Float16* __restrict__ hseq,      // ws [B,T,HID] f16
    float* __restrict__ hT_out)       // [B,HID] f32
{
    __shared__ _Float16 lds_h[16][16];

    const int lane = threadIdx.x & 31;
    const int ln = lane & 15;
    const int hl = lane >> 4;
    const int b0 = blockIdx.x * 16;

    // ---- Preload W_ih^T and W_hh^T as f16 B-tiles (K in 32-chunks) ----
    // B layout (16-bit, 32x16): lane holds (k = hl*8 + h + (h & 8), n = ln), h=0..15
    v16h Bx[3][2], Bh[3];
#pragma unroll
    for (int nt = 0; nt < 3; ++nt) {
        const int row = nt * 16 + ln;      // gate-output index n
#pragma unroll
        for (int c = 0; c < 2; ++c) {
            v16h bb;
#pragma unroll
            for (int h = 0; h < 16; ++h) {
                int k = c * 32 + hl * 8 + h + (h & 8);
                float w = (k < OBS) ? W_ih[row * OBS + k] : 0.0f;
                bb[h] = (_Float16)w;
            }
            Bx[nt][c] = bb;
        }
        v16h bh;
#pragma unroll
        for (int h = 0; h < 16; ++h) {
            int k = hl * 8 + h + (h & 8);
            float w = (k < HID) ? W_hh[row * HID + k] : 0.0f;
            bh[h] = (_Float16)w;
        }
        Bh[nt] = bh;
    }

    // Per-lane biases (C-layout column n = ln within each gate tile)
    const float bir = b_ih[ln],      bhr = b_hh[ln];
    const float biz = b_ih[16 + ln], bhz = b_hh[16 + ln];
    const float bin = b_ih[32 + ln], bhn = b_hh[32 + ln];

    // h in f32, C-layout: element i -> (m = i + 8*hl, n = ln)
    float h[8];
#pragma unroll
    for (int i = 0; i < 8; ++i)
        h[i] = h0[(size_t)(b0 + i + 8 * hl) * HID + ln];
    // seed LDS with h_0 (f16, row-major [m][k])
#pragma unroll
    for (int i = 0; i < 8; ++i)
        lds_h[i + 8 * hl][ln] = (_Float16)h[i];

    // A-row base for this lane (m = ln)
    const float* xrow = x + (size_t)(b0 + ln) * TLEN * OBS;
    _Float16* hseq_row = hseq + (size_t)(b0 + ln) * TLEN * HID + hl * 8;

    for (int t = 0; t < TLEN; ++t) {
        // ---- x A-tiles for this timestep (K=48 -> 2 chunks of 32); h-independent
        const float* xt = xrow + (size_t)t * OBS;
        v4f x0 = *(const v4f*)(xt + hl * 8);
        v4f x1 = *(const v4f*)(xt + hl * 8 + 4);
        v4f x2 = *(const v4f*)(xt + 16 + hl * 8);
        v4f x3 = *(const v4f*)(xt + 16 + hl * 8 + 4);
        v4f x4 = *(const v4f*)(xt + 32 + hl * 8);
        v4f x5 = *(const v4f*)(xt + 32 + hl * 8 + 4);
        v16h a0, a1;
#pragma unroll
        for (int j = 0; j < 4; ++j) {
            a0[j]      = (_Float16)x0[j];  a0[4 + j]  = (_Float16)x1[j];
            a0[8 + j]  = (_Float16)x2[j];  a0[12 + j] = (_Float16)x3[j];
            a1[j]      = (_Float16)x4[j];  a1[4 + j]  = (_Float16)x5[j];
            a1[8 + j]  = (_Float16)0;      a1[12 + j] = (_Float16)0;
        }

        // ---- 6 x-projection WMMAs (independent of h)
        v8f accR = {}, accZ = {}, accXN = {}, accHN = {};
        accR  = wmma_f16(a0, Bx[0][0], accR);
        accR  = wmma_f16(a1, Bx[0][1], accR);
        accZ  = wmma_f16(a0, Bx[1][0], accZ);
        accZ  = wmma_f16(a1, Bx[1][1], accZ);
        accXN = wmma_f16(a0, Bx[2][0], accXN);
        accXN = wmma_f16(a1, Bx[2][1], accXN);

        // ---- h_t A-row from LDS (in-order DS vs. prev-iter scatter stores)
        v8h hrow = *(const v8h*)(&lds_h[ln][hl * 8]);
        // emit h_seq[t-1] (= h_t) as a single 16B store per lane
        if (t > 0)
            *(v8h*)(hseq_row + (size_t)(t - 1) * HID) = hrow;
        v16h hA;
#pragma unroll
        for (int j = 0; j < 8; ++j) { hA[j] = hrow[j]; hA[8 + j] = (_Float16)0; }

        // ---- 3 h-projection WMMAs (the recurrence chain)
        accR  = wmma_f16(hA, Bh[0], accR);
        accZ  = wmma_f16(hA, Bh[1], accZ);
        accHN = wmma_f16(hA, Bh[2], accHN);

        // ---- gates + state update (f32, element-wise in C-layout),
        //      then scatter h_{t+1} back to LDS for the next step
#pragma unroll
        for (int i = 0; i < 8; ++i) {
            float r = sigmoidf_fast(accR[i] + bir + bhr);
            float z = sigmoidf_fast(accZ[i] + biz + bhz);
            float n = tanhf_fast(accXN[i] + bin + r * (accHN[i] + bhn));
            h[i] = (1.0f - z) * n + z * h[i];
            lds_h[i + 8 * hl][ln] = (_Float16)h[i];
        }
    }

    // flush last h_seq row (h_T) and final hidden state (f32)
    {
        v8h hrow = *(const v8h*)(&lds_h[ln][hl * 8]);
        *(v8h*)(hseq_row + (size_t)(TLEN - 1) * HID) = hrow;
    }
#pragma unroll
    for (int i = 0; i < 8; ++i)
        hT_out[(size_t)(b0 + i + 8 * hl) * HID + ln] = h[i];
}

// ---------------------------------------------------------------------------
// Kernel 2: token-parallel MLP + heads. 8 waves/block, 16-token tiles/wave.
// W1 & head weights staged in LDS (B-tile order), W2 held in VGPRs.
// ---------------------------------------------------------------------------
__global__ __launch_bounds__(256) void mlp_head_kernel(
    const float* __restrict__ x,         // [B,T,OBS]
    const _Float16* __restrict__ hseq,   // [B,T,HID] f16
    const float* __restrict__ W1, const float* __restrict__ b1,   // [128,64],[128]
    const float* __restrict__ W2, const float* __restrict__ b2,   // [128,128],[128]
    const float* __restrict__ W_mu, const float* __restrict__ b_mu, // [12,128],[12]
    const float* __restrict__ W_v,  const float* __restrict__ b_v,  // [1,128],[1]
    float* __restrict__ mu, float* __restrict__ vout)
{
    __shared__ _Float16 sW1[8 * 2 * 32 * 16];  // 16 KB: W1^T B-tiles [nt][c][lane][h]
    __shared__ _Float16 sWh[4 * 32 * 16];      //  4 KB: head B-tiles [c][lane][h]
    __shared__ _Float16 sZ[8 * 16 * 128];      // 32 KB: per-wave z scratch [16][128]

    const int tid = threadIdx.x;
    const int lane = tid & 31;
    const int warp = tid >> 5;
    const int ln = lane & 15;
    const int hl = lane >> 4;

    // ---- cooperative staging of W1^T (K=64 -> 2 chunks) into LDS
    for (int f = tid; f < 8 * 2 * 32 * 16; f += 256) {
        int h16 = f & 15;
        int l   = (f >> 4) & 31;
        int c   = (f >> 9) & 1;
        int nt  = f >> 10;
        int n = nt * 16 + (l & 15);
        int k = c * 32 + (l >> 4) * 8 + h16 + (h16 & 8);      // < 64 always
        sW1[f] = (_Float16)W1[n * 64 + k];
    }
    // ---- staging of concatenated head [mu;v;pad] (16x128) B-tiles
    for (int f = tid; f < 4 * 32 * 16; f += 256) {
        int h16 = f & 15;
        int l   = (f >> 4) & 31;
        int c   = f >> 9;
        int n = l & 15;
        int k = c * 32 + (l >> 4) * 8 + h16 + (h16 & 8);      // < 128 always
        float w = (n < ACTD) ? W_mu[n * H2 + k] : ((n == ACTD) ? W_v[k] : 0.0f);
        sWh[f] = (_Float16)w;
    }
    __syncthreads();

    // ---- per-lane biases
    float b1v[8], b2v[8];
#pragma unroll
    for (int nt = 0; nt < 8; ++nt) {
        b1v[nt] = b1[nt * 16 + ln];
        b2v[nt] = b2[nt * 16 + ln];
    }
    const float bh = (ln < ACTD) ? b_mu[ln] : ((ln == ACTD) ? b_v[0] : 0.0f);

    // ---- W2^T B-tiles in registers (8 N-tiles x 4 K-chunks)
    v16h W2B[8][4];
#pragma unroll
    for (int nt = 0; nt < 8; ++nt) {
#pragma unroll
        for (int c = 0; c < 4; ++c) {
            const float* wp = W2 + (nt * 16 + ln) * H1 + c * 32 + hl * 8;
            v4f w0 = *(const v4f*)(wp);
            v4f w1 = *(const v4f*)(wp + 4);
            v4f w2 = *(const v4f*)(wp + 16);
            v4f w3 = *(const v4f*)(wp + 20);
            v16h bb;
#pragma unroll
            for (int j = 0; j < 4; ++j) {
                bb[j]      = (_Float16)w0[j];  bb[4 + j]  = (_Float16)w1[j];
                bb[8 + j]  = (_Float16)w2[j];  bb[12 + j] = (_Float16)w3[j];
            }
            W2B[nt][c] = bb;
        }
    }

    _Float16* zs = sZ + warp * 16 * 128;

    const int NT = (BATCH * TLEN) / 16;          // 32768 token tiles
    const int nwaves = gridDim.x * 8;
    for (int tile = blockIdx.x * 8 + warp; tile < NT; tile += nwaves) {
        const size_t tokm = (size_t)tile * 16 + ln;   // A-row m = ln

        // ---- hx = [h(16) | x(48)] A-tiles, K=64 -> 2 chunks
        v8h hv = *(const v8h*)(hseq + tokm * HID + hl * 8);
        const float* xp = x + tokm * OBS;
        v4f xa = *(const v4f*)(xp + hl * 8);
        v4f xb = *(const v4f*)(xp + hl * 8 + 4);
        v4f xc = *(const v4f*)(xp + 16 + hl * 8);
        v4f xd = *(const v4f*)(xp + 16 + hl * 8 + 4);
        v4f xe = *(const v4f*)(xp + 32 + hl * 8);
        v4f xf = *(const v4f*)(xp + 32 + hl * 8 + 4);
        v16h a0, a1;
#pragma unroll
        for (int j = 0; j < 4; ++j) {
            a0[j]      = hv[j];             a0[4 + j]  = hv[4 + j];
            a0[8 + j]  = (_Float16)xa[j];   a0[12 + j] = (_Float16)xb[j];
            a1[j]      = (_Float16)xc[j];   a1[4 + j]  = (_Float16)xd[j];
            a1[8 + j]  = (_Float16)xe[j];   a1[12 + j] = (_Float16)xf[j];
        }

        // ---- layer 1: z1 = tanh(hx @ W1^T + b1), 16 WMMAs
#pragma unroll
        for (int nt = 0; nt < 8; ++nt) {
            v8f acc = {};
            acc = wmma_f16(a0, *(const v16h*)(sW1 + ((nt * 2 + 0) * 32 + lane) * 16), acc);
            acc = wmma_f16(a1, *(const v16h*)(sW1 + ((nt * 2 + 1) * 32 + lane) * 16), acc);
#pragma unroll
            for (int i = 0; i < 8; ++i)
                zs[(i + 8 * hl) * 128 + nt * 16 + ln] = (_Float16)tanhf_fast(acc[i] + b1v[nt]);
        }
        // in-order DS per wave: relayout loads below see the stores above
        v16h zA[4];
#pragma unroll
        for (int c = 0; c < 4; ++c) {
            v8h lo = *(const v8h*)(zs + ln * 128 + c * 32 + hl * 8);
            v8h hi = *(const v8h*)(zs + ln * 128 + c * 32 + 16 + hl * 8);
            v16h tv;
#pragma unroll
            for (int j = 0; j < 8; ++j) { tv[j] = lo[j]; tv[8 + j] = hi[j]; }
            zA[c] = tv;
        }

        // ---- layer 2: z2 = tanh(z1 @ W2^T + b2), 32 WMMAs
#pragma unroll
        for (int nt = 0; nt < 8; ++nt) {
            v8f acc = {};
#pragma unroll
            for (int c = 0; c < 4; ++c)
                acc = wmma_f16(zA[c], W2B[nt][c], acc);
#pragma unroll
            for (int i = 0; i < 8; ++i)
                zs[(i + 8 * hl) * 128 + nt * 16 + ln] = (_Float16)tanhf_fast(acc[i] + b2v[nt]);
        }
        v16h z2A[4];
#pragma unroll
        for (int c = 0; c < 4; ++c) {
            v8h lo = *(const v8h*)(zs + ln * 128 + c * 32 + hl * 8);
            v8h hi = *(const v8h*)(zs + ln * 128 + c * 32 + 16 + hl * 8);
            v16h tv;
#pragma unroll
            for (int j = 0; j < 8; ++j) { tv[j] = lo[j]; tv[8 + j] = hi[j]; }
            z2A[c] = tv;
        }

        // ---- heads: [mu(12); v(1); pad(3)] = z2 @ Whead^T, 4 WMMAs
        v8f acc = {};
#pragma unroll
        for (int c = 0; c < 4; ++c)
            acc = wmma_f16(z2A[c], *(const v16h*)(sWh + (c * 32 + lane) * 16), acc);
#pragma unroll
        for (int i = 0; i < 8; ++i) {
            float o = acc[i] + bh;
            size_t tok = (size_t)tile * 16 + i + 8 * hl;
            if (ln < ACTD)       mu[tok * ACTD + ln] = o;
            else if (ln == ACTD) vout[tok] = o;
        }
    }
}

// ---------------------------------------------------------------------------
extern "C" void kernel_launch(void* const* d_in, const int* in_sizes, int n_in,
                              void* d_out, int out_size, void* d_ws, size_t ws_size,
                              hipStream_t stream) {
    const float* x    = (const float*)d_in[0];
    const float* h0   = (const float*)d_in[1];   // [1,2048,16]
    const float* W_ih = (const float*)d_in[2];
    const float* W_hh = (const float*)d_in[3];
    const float* b_ih = (const float*)d_in[4];
    const float* b_hh = (const float*)d_in[5];
    const float* W1   = (const float*)d_in[6];
    const float* b1   = (const float*)d_in[7];
    const float* W2   = (const float*)d_in[8];
    const float* b2   = (const float*)d_in[9];
    const float* W_mu = (const float*)d_in[10];
    const float* b_mu = (const float*)d_in[11];
    const float* W_v  = (const float*)d_in[12];
    const float* b_v  = (const float*)d_in[13];

    float* out = (float*)d_out;
    float* mu  = out;                                        // [B,T,12]
    float* vv  = out + (size_t)BATCH * TLEN * ACTD;          // [B,T]
    float* hT  = vv  + (size_t)BATCH * TLEN;                 // [B,16]

    _Float16* hseq = (_Float16*)d_ws;                        // [B,T,16] f16 (16 MB)

    gru_scan_kernel<<<BATCH / 16, 32, 0, stream>>>(
        x, h0, W_ih, W_hh, b_ih, b_hh, hseq, hT);

    mlp_head_kernel<<<1024, 256, 0, stream>>>(
        x, hseq, W1, b1, W2, b2, W_mu, b_mu, W_v, b_v, mu, vv);

    (void)in_sizes; (void)n_in; (void)out_size; (void)ws_size;
}
